// WindowAttention_84980222918995
// MI455X (gfx1250) — matrix operand order repaired and checked by reading, not compiled, not run on hardware
//
#include <hip/hip_runtime.h>

typedef __attribute__((ext_vector_type(16))) _Float16 v16h;
typedef __attribute__((ext_vector_type(8)))  _Float16 v8h;
typedef __attribute__((ext_vector_type(4)))  _Float16 v4h;
typedef __attribute__((ext_vector_type(8)))  float    v8f;
typedef __attribute__((ext_vector_type(4)))  float    v4f;

#define NTOK   49
#define CDIM   128
#define NHEAD  4
#define HDIM   32
#define NPAD   64

// d_ws layout (bytes):
//   [0,      98304)  qkv_w as f16   (384x128)
//   [98304, 131072)  proj_w as f16  (128x128)
//   [131072,169488)  bias[h][n][m]  f32 (4x49x49)
#define WS_WQKV  0
#define WS_WPROJ 98304
#define WS_BIAS  131072

// LDS layout (bytes), 128 KB total:
//   [0,      32768)  sQKV : 64 x 256 f16   (q cols 0..127 pre-scaled, k cols 128..255)
//   [32768,  49152)  sVT  : [h][32][64] f16 (v transposed: dd x m)
//   [49152, 114688)  sAttn: [h][64][64] f32, reused in-place as f16 rows (pitch 256B)
//                    sX   : 64 x 128 f16 aliases first 16 KB (dead after QKV GEMM)
//   [114688,131072)  sOut : 64 x 128 f16
#define LDS_BYTES 131072

__global__ void winattn_prep_kernel(const float* __restrict__ qkv_w,
                                    const float* __restrict__ proj_w,
                                    const float* __restrict__ bias_table,
                                    const int*   __restrict__ rel_index,
                                    _Float16* __restrict__ wqkv,
                                    _Float16* __restrict__ wproj,
                                    float*    __restrict__ biasf) {
  int t = blockIdx.x * 256 + threadIdx.x;
  if (t < 384 * 128) wqkv[t]  = (_Float16)qkv_w[t];
  if (t < 128 * 128) wproj[t] = (_Float16)proj_w[t];
  if (t < NHEAD * NTOK * NTOK) {
    int h = t / (NTOK * NTOK);
    int nm = t % (NTOK * NTOK);
    biasf[t] = bias_table[rel_index[nm] * NHEAD + h];
  }
}

__device__ __forceinline__ v16h make_frag(const v8h* p0, const v8h* p1) {
  union { v16h v; v8h h2[2]; } u;
  u.h2[0] = *p0;
  u.h2[1] = *p1;
  return u.v;
}

__global__ __launch_bounds__(256)
void winattn_main_kernel(const float* __restrict__ x,
                         const float* __restrict__ mask,
                         const float* __restrict__ qkv_b,
                         const float* __restrict__ proj_b,
                         const _Float16* __restrict__ wqkv,
                         const _Float16* __restrict__ wproj,
                         const float* __restrict__ biasf,
                         float* __restrict__ out,
                         int nW) {
  __shared__ __align__(16) unsigned char lds[LDS_BYTES];
  _Float16* sQKV  = (_Float16*)(lds);
  _Float16* sVT   = (_Float16*)(lds + 32768);
  float*    sAttn = (float*)(lds + 49152);
  _Float16* sX    = (_Float16*)(lds + 49152);
  _Float16* sOut  = (_Float16*)(lds + 114688);

  const int tid  = threadIdx.x;
  const int lane = tid & 31;
  const int wvu  = __builtin_amdgcn_readfirstlane(tid >> 5);  // scalar wave id
  const int half = lane >> 4;   // 0: lanes 0-15, 1: lanes 16-31
  const int l16  = lane & 15;
  const int b    = blockIdx.x;
  const int w    = b % nW;
  const float qscale = 0.17677669529663687f;  // 32^-0.5

  // ---------------- Phase 0: x -> sX (f16, pad rows 49..63 zero) ----------------
  const v4f* xb4 = (const v4f*)(x + (size_t)b * (NTOK * CDIM));
  for (int i = tid; i < (NPAD * CDIM) / 4; i += 256) {  // 2048 quads
    v4h v = {};
    if (i < (NTOK * CDIM) / 4) {
      v4f f = xb4[i];
      v[0] = (_Float16)f[0]; v[1] = (_Float16)f[1];
      v[2] = (_Float16)f[2]; v[3] = (_Float16)f[3];
    }
    *(v4h*)(sX + i * 4) = v;
  }
  __syncthreads();

  // ---------------- Phase 1: QKV = X @ Wqkv^T + b -------------------------------
  // Wave owns fixed mt; loops 12 nt tiles; A fragments preloaded once.
  {
    int mt  = wvu & 3;
    int grp = wvu >> 2;                  // 0: nt 0..11, 1: nt 12..23
    int arow = mt * 16 + l16;
    v16h af[4];
    for (int kk = 0; kk < 4; ++kk) {
      int ka = kk * 32 + half * 8;
      af[kk] = make_frag((const v8h*)(sX + arow * CDIM + ka),
                         (const v8h*)(sX + arow * CDIM + ka + 16));
    }
    for (int ntl = 0; ntl < 12; ++ntl) {
      int nt = grp * 12 + ntl;
      int brow = nt * 16 + l16;
      v8f acc = {};
      for (int kk = 0; kk < 4; ++kk) {
        int kb = kk * 32 + half * 8;
        v16h bm = make_frag((const v8h*)(wqkv + brow * CDIM + kb),
                            (const v8h*)(wqkv + brow * CDIM + kb + 16));
        acc = __builtin_amdgcn_wmma_f32_16x16x32_f16(false, af[kk], false, bm,
                                                     (short)0, acc, false, false);
      }
      int j = nt * 16 + l16;
      float bj = qkv_b[j];
      if (nt < 8) {                                      // q section, pre-scaled
        for (int r = 0; r < 8; ++r) {
          int m = mt * 16 + r + half * 8;
          sQKV[m * 256 + j] = (_Float16)((acc[r] + bj) * qscale);
        }
      } else if (nt < 16) {                              // k section
        for (int r = 0; r < 8; ++r) {
          int m = mt * 16 + r + half * 8;
          sQKV[m * 256 + j] = (_Float16)(acc[r] + bj);
        }
      } else {                                           // v -> transposed store
        int jh = j - 256;
        int h = jh >> 5, dd = jh & 31;
        v8h pk;
        for (int r = 0; r < 8; ++r) pk[r] = (_Float16)(acc[r] + bj);
        *(v8h*)(sVT + (h * 32 + dd) * 64 + mt * 16 + half * 8) = pk;
      }
    }
  }
  __syncthreads();

  // ---------------- Phase 2: attn = q@k^T + bias + mask (f32) -------------------
  // Wave owns (head, mt-pair); preloads 2 q-frags + 4 k-frags.
  {
    int h   = wvu >> 1;
    int mt0 = (wvu & 1) * 2;
    int ka  = h * 32 + half * 8;
    v16h aq[2];
    for (int mi = 0; mi < 2; ++mi) {
      int qrow = (mt0 + mi) * 16 + l16;
      aq[mi] = make_frag((const v8h*)(sQKV + qrow * 256 + ka),
                         (const v8h*)(sQKV + qrow * 256 + ka + 16));
    }
    int kb = 128 + h * 32 + half * 8;
    v16h bk[4];
    for (int nt = 0; nt < 4; ++nt) {
      int krow = nt * 16 + l16;
      bk[nt] = make_frag((const v8h*)(sQKV + krow * 256 + kb),
                         (const v8h*)(sQKV + krow * 256 + kb + 16));
    }
    const float* biasH = biasf + h * (NTOK * NTOK);
    const float* maskW = mask + (size_t)w * (NTOK * NTOK);
    for (int mi = 0; mi < 2; ++mi) {
      int mt = mt0 + mi;
      for (int nt = 0; nt < 4; ++nt) {
        v8f acc = {};
        acc = __builtin_amdgcn_wmma_f32_16x16x32_f16(false, aq[mi], false, bk[nt],
                                                     (short)0, acc, false, false);
        int n_  = nt * 16 + l16;
        int ncl = n_ < 48 ? n_ : 48;
        for (int r = 0; r < 8; ++r) {
          int m   = mt * 16 + r + half * 8;
          int mcl = m < 48 ? m : 48;
          float add = biasH[mcl * NTOK + ncl] + maskW[mcl * NTOK + ncl];
          float vv  = (m < NTOK && n_ < NTOK) ? (acc[r] + add) : -1e30f;
          sAttn[h * 4096 + m * 64 + n_] = vv;
        }
      }
    }
  }
  __syncthreads();

  // ---------------- Phase 3: row softmax, f32 -> f16 in place -------------------
  if (tid < NHEAD * NTOK) {
    int h = tid / NTOK, i = tid % NTOK;
    unsigned char* rowB = lds + 49152 + h * 16384 + i * 256;
    const v4f* rf4 = (const v4f*)rowB;
    float mx = -1e30f;
    for (int c4 = 0; c4 < 16; ++c4) {
      v4f v = rf4[c4];
      mx = fmaxf(mx, fmaxf(fmaxf(v[0], v[1]), fmaxf(v[2], v[3])));
    }
    volatile float*    rf = (volatile float*)rowB;
    volatile _Float16* rh = (volatile _Float16*)rowB;
    float sum = 0.0f;
    for (int t = 0; t < 32; ++t) {           // pairwise: f16 writes clobber only
      float v0 = rf[2 * t];                  // already-consumed f32 words
      float v1 = rf[2 * t + 1];
      float e0 = __expf(v0 - mx);
      float e1 = __expf(v1 - mx);
      sum += e0 + e1;
      rh[2 * t]     = (_Float16)e0;
      rh[2 * t + 1] = (_Float16)e1;
    }
    float inv = 1.0f / sum;
    v8h* r8 = (v8h*)rowB;
    for (int q8 = 0; q8 < 8; ++q8) {
      v8h e = r8[q8];
      for (int k = 0; k < 8; ++k) e[k] = (_Float16)((float)e[k] * inv);
      r8[q8] = e;
    }
  }
  __syncthreads();

  // ---------------- Phase 4: out_h = attn @ v -----------------------------------
  // Wave owns (head, nt); preloads 2 v^T fragments, reused across 4 mt.
  {
    int h  = wvu >> 1;
    int nt = wvu & 1;
    int bn = nt * 16 + l16;
    v16h bv[2];
    for (int kk = 0; kk < 2; ++kk) {
      int kb = kk * 32 + half * 8;
      bv[kk] = make_frag((const v8h*)(sVT + (h * 32 + bn) * 64 + kb),
                         (const v8h*)(sVT + (h * 32 + bn) * 64 + kb + 16));
    }
    const _Float16* attnH = (const _Float16*)(lds + 49152 + h * 16384); // pitch 128
    int col = h * 32 + nt * 16 + l16;
    for (int mt = 0; mt < 4; ++mt) {
      int arow = mt * 16 + l16;
      v8f acc = {};
      for (int kk = 0; kk < 2; ++kk) {
        int ka = kk * 32 + half * 8;
        v16h a = make_frag((const v8h*)(attnH + arow * 128 + ka),
                           (const v8h*)(attnH + arow * 128 + ka + 16));
        acc = __builtin_amdgcn_wmma_f32_16x16x32_f16(false, a, false, bv[kk],
                                                     (short)0, acc, false, false);
      }
      for (int r = 0; r < 8; ++r) {
        int i = mt * 16 + r + half * 8;
        sOut[i * CDIM + col] = (_Float16)acc[r];
      }
    }
  }
  __syncthreads();

  // ---------------- Phase 5: out = sOut @ Wproj^T + b -> global -----------------
  // Wave owns fixed mt; A fragments preloaded, reused across 4 nt.
  {
    int mt  = wvu & 3;
    int grp = wvu >> 2;                 // 0: nt 0..3, 1: nt 4..7
    int arow = mt * 16 + l16;
    v16h af[4];
    for (int kk = 0; kk < 4; ++kk) {
      int ka = kk * 32 + half * 8;
      af[kk] = make_frag((const v8h*)(sOut + arow * CDIM + ka),
                         (const v8h*)(sOut + arow * CDIM + ka + 16));
    }
    float* outb = out + (size_t)b * (NTOK * CDIM);
    for (int ntl = 0; ntl < 4; ++ntl) {
      int nt = grp * 4 + ntl;
      int brow = nt * 16 + l16;
      v8f acc = {};
      for (int kk = 0; kk < 4; ++kk) {
        int kb = kk * 32 + half * 8;
        v16h bm = make_frag((const v8h*)(wproj + brow * CDIM + kb),
                            (const v8h*)(wproj + brow * CDIM + kb + 16));
        acc = __builtin_amdgcn_wmma_f32_16x16x32_f16(false, af[kk], false, bm,
                                                     (short)0, acc, false, false);
      }
      int j = nt * 16 + l16;
      float bj = proj_b[j];
      if (mt < 3) {                                // rows all < 48: unconditional
        for (int r = 0; r < 8; ++r) {
          int i = mt * 16 + r + half * 8;
          outb[i * CDIM + j] = acc[r] + bj;
        }
      } else {                                     // mt==3: only row 48 valid
        if (half == 0) outb[48 * CDIM + j] = acc[0] + bj;
      }
    }
  }
}

extern "C" void kernel_launch(void* const* d_in, const int* in_sizes, int n_in,
                              void* d_out, int out_size, void* d_ws, size_t ws_size,
                              hipStream_t stream) {
  const float* x          = (const float*)d_in[0];
  const float* mask       = (const float*)d_in[1];
  const float* qkv_w      = (const float*)d_in[2];
  const float* qkv_b      = (const float*)d_in[3];
  const float* proj_w     = (const float*)d_in[4];
  const float* proj_b     = (const float*)d_in[5];
  const float* bias_table = (const float*)d_in[6];
  const int*   rel_index  = (const int*)d_in[7];

  int B_ = in_sizes[0] / (NTOK * CDIM);
  int nW = in_sizes[1] / (NTOK * NTOK);

  _Float16* wqkv  = (_Float16*)((char*)d_ws + WS_WQKV);
  _Float16* wproj = (_Float16*)((char*)d_ws + WS_WPROJ);
  float*    biasf = (float*)((char*)d_ws + WS_BIAS);

  winattn_prep_kernel<<<(384 * 128 + 255) / 256, 256, 0, stream>>>(
      qkv_w, proj_w, bias_table, rel_index, wqkv, wproj, biasf);

  winattn_main_kernel<<<B_, 256, 0, stream>>>(
      x, mask, qkv_b, proj_b, wqkv, wproj, biasf, (float*)d_out, nW);
}
